// Alembic_Layer_30923764531483
// MI455X (gfx1250) — compile-verified, matrix operand order checked by reading, not run on hardware
//
#include <hip/hip_runtime.h>

typedef _Float16 v8h  __attribute__((ext_vector_type(8)));
typedef _Float16 v16h __attribute__((ext_vector_type(16)));
typedef float    v8f  __attribute__((ext_vector_type(8)));

#define CC 32
#define LL 500
#define KK 101
#define FF 3
#define XS_STRIDE 648       // f16 elements per LDS row (1296 B = 324 dwords; 324 % 64 == 4)
#define XPAD 50             // left zero padding in LDS row
#define TAPS_STRIDE 160     // f16 per filter (16 lead zeros + 101 taps + tail zeros)

__global__ __launch_bounds__(256)
void fir_depthwise_wmma(const float* __restrict__ x,
                        const float* __restrict__ params,
                        float* __restrict__ out) {
    __shared__ __align__(16) _Float16 xs[CC * XS_STRIDE];       // 41472 B
    __shared__ __align__(16) _Float16 taps[FF * TAPS_STRIDE];   // 960 B

    const int b    = blockIdx.x;
    const int tid  = threadIdx.x;
    const int lane = tid & 31;
    const int wave = tid >> 5;
    const int n    = lane & 15;   // N index (output position within tile)
    const int grp  = lane >> 4;   // lane half-group

    // ---------- Phase 1a: build f16 taps (windowed sinc), zero-padded ----------
    const float PIF = 3.14159265358979f;
    for (int idx = tid; idx < FF * TAPS_STRIDE; idx += 256) {
        int f = idx / TAPS_STRIDE;
        int k = (idx - f * TAPS_STRIDE) - 16;
        float tap = 0.0f;
        if (k >= 0 && k < KK) {
            float lo = params[(b * FF + f) * 2 + 0] * (1.0f / 64.0f);  // /nyq
            float hi = params[(b * FF + f) * 2 + 1] * (1.0f / 64.0f);
            float nn = (float)k - 50.0f;
            if (k == 50) tap = hi - lo;
            else         tap = (__sinf(PIF * nn * hi) - __sinf(PIF * nn * lo)) / (PIF * nn);
            float win = 0.5f - 0.5f * __cosf(2.0f * PIF * (float)k * (1.0f / 101.0f));
            tap *= win;
        }
        taps[idx] = (_Float16)tap;
    }

    // ---------- Phase 1b: stage x[b] -> LDS f16 with +-50 zero padding ----------
    {
        const float* xb = x + (size_t)b * CC * LL;
        unsigned int* xsw = (unsigned int*)xs;
        const int dw_per_row = XS_STRIDE / 2;   // 324 dwords per row
        for (int p = tid; p < CC * dw_per_row; p += 256) {
            int row = p / dw_per_row;
            int pp  = p - row * dw_per_row;
            int j0  = 2 * pp - XPAD;
            int j1  = j0 + 1;
            _Float16 h0 = (j0 >= 0 && j0 < LL) ? (_Float16)xb[row * LL + j0] : (_Float16)0.0f;
            _Float16 h1 = (j1 >= 0 && j1 < LL) ? (_Float16)xb[row * LL + j1] : (_Float16)0.0f;
            unsigned short u0 = __builtin_bit_cast(unsigned short, h0);
            unsigned short u1 = __builtin_bit_cast(unsigned short, h1);
            xsw[p] = (unsigned int)u0 | ((unsigned int)u1 << 16);
        }
    }
    __syncthreads();

    // ---------- Phase 2: WMMA tiles ----------
    for (int f = 0; f < FF; ++f) {
        // Toeplitz B fragments: B[k,n] = taps[k-n], 4 chunks of K=32
        v16h bfrag[4];
#pragma unroll
        for (int kk = 0; kk < 4; ++kk) {
#pragma unroll
            for (int h = 0; h < 16; ++h) {
                int k = 32 * kk + 16 * (h >> 3) + 8 * grp + (h & 7);
                bfrag[kk][h] = taps[f * TAPS_STRIDE + 16 + k - n];
            }
        }

        // 64 tiles (2 channel-tiles x 32 position-tiles), 8 per wave
        for (int t = 0; t < 8; ++t) {
            int tile   = wave * 8 + t;
            int c_tile = tile & 1;
            int l_tile = tile >> 1;

            const _Float16* arow = xs + (c_tile * 16 + (lane & 15)) * XS_STRIDE;
            v8f acc = {};
#pragma unroll
            for (int kk = 0; kk < 4; ++kk) {
                int s1 = 16 * l_tile + 32 * kk + 8 * grp;      // multiple of 8 -> 16B aligned
                v8h lov = *(const v8h*)(arow + s1);
                v8h hiv = *(const v8h*)(arow + s1 + 16);
                v16h a;
#pragma unroll
                for (int i = 0; i < 8; ++i) { a[i] = lov[i]; a[i + 8] = hiv[i]; }
                acc = __builtin_amdgcn_wmma_f32_16x16x32_f16(
                        false, a, false, bfrag[kk], (short)0, acc, false, false);
            }

            // Store D: VGPR v -> channel (c_tile*16 + v + 8*grp), position 16*l_tile + n
            int l = 16 * l_tile + n;
            if (l < LL) {
                float* ob = out + (((size_t)b * FF + f) * CC + c_tile * 16 + 8 * grp) * LL + l;
#pragma unroll
                for (int v = 0; v < 8; ++v) {
                    ob[(size_t)v * LL] = acc[v];
                }
            }
        }
    }
}

extern "C" void kernel_launch(void* const* d_in, const int* in_sizes, int n_in,
                              void* d_out, int out_size, void* d_ws, size_t ws_size,
                              hipStream_t stream) {
    const float* x      = (const float*)d_in[0];   // (512*32, 500) fp32
    const float* params = (const float*)d_in[1];   // (512, 3, 2) fp32
    float* out          = (float*)d_out;           // (512, 3, 32, 500) fp32
    (void)in_sizes; (void)n_in; (void)out_size; (void)d_ws; (void)ws_size;

    fir_depthwise_wmma<<<512, 256, 0, stream>>>(x, params, out);
}
